// Encoder_79645873537161
// MI455X (gfx1250) — matrix operand order repaired and checked by reading, not compiled, not run on hardware
//
#include <hip/hip_runtime.h>
#include <hip/hip_bf16.h>
#include <math.h>

// ---------------------------------------------------------------------------
// 2-layer GRU (Keras reset_after=True, gates [z,r,h]) for MI455X/gfx1250.
//
// ~412 GFLOP total, ~0.8 GB HBM (~35us @ 23.3 TB/s). Input projections: two
// [16384x1024]x[1024x3072] GEMMs on v_wmma_f32_16x16x32_bf16 (f32 accum),
// 128x128 block tiles, 32x64 wave tiles, manual even/odd register-set
// pipeline (no rotate-copies). Recurrent scan: ONE persistent kernel per
// layer — 64 blocks, each with its 48x1024 bf16 U-slice resident in LDS
// (96KB of 320KB/WGP), looping 256 timesteps with a device-scope atomic
// grid barrier per step. No per-step launch overhead; U read once.
// Carried state / XP stay f32; matrix operands use bf16 shadows.
// ---------------------------------------------------------------------------

typedef __bf16 bf16_t;
typedef __attribute__((ext_vector_type(16))) __bf16 bf16x16;
typedef __attribute__((ext_vector_type(8)))  __bf16 bf16x8;
typedef __attribute__((ext_vector_type(8)))  float  f32x8;

#define B_N  64
#define T_N  256
#define F_N  1024
#define U_N  1024
#define G_N  3072

__device__ __forceinline__ f32x8 wmma_bf16(bf16x16 a, bf16x16 b, f32x8 c) {
  return __builtin_amdgcn_wmma_f32_16x16x32_bf16(false, a, false, b,
                                                 (short)0, c, false, false);
}

// A fragment (16x32 bf16) from a row-major source.
// p = &A[(m0 + (lane&15))*ld + k0 + (lane>>4)*16].
__device__ __forceinline__ bf16x16 load_a_frag(const bf16_t* p) {
  bf16x8 lo = *(const bf16x8*)(p);
  bf16x8 hi = *(const bf16x8*)(p + 8);
  bf16x16 f;
#pragma unroll
  for (int i = 0; i < 8; ++i) { f[i] = lo[i]; f[i + 8] = hi[i]; }
  return f;
}

__device__ __forceinline__ float sigmoidf(float x) {
  return 1.0f / (1.0f + expf(-x));
}

// ---------------- prep kernels -------------------------------------------

// W [K,N] f32 row-major -> Wt [N,K] bf16 (row n of Wt = column n of W).
__global__ void __launch_bounds__(256)
transpose_cvt(const float* __restrict__ W, bf16_t* __restrict__ Wt,
              int K, int N) {
  __shared__ float tile[32][33];
  int n0 = blockIdx.x * 32, k0 = blockIdx.y * 32;
  int tx = threadIdx.x, ty = threadIdx.y;       // block (32, 8)
#pragma unroll
  for (int i = 0; i < 4; ++i)
    tile[ty + 8 * i][tx] = W[(size_t)(k0 + ty + 8 * i) * N + (n0 + tx)];
  __syncthreads();
#pragma unroll
  for (int i = 0; i < 4; ++i)
    Wt[(size_t)(n0 + ty + 8 * i) * K + (k0 + tx)] =
        (bf16_t)tile[tx][ty + 8 * i];
}

__global__ void __launch_bounds__(256)
cvt_bf16(const float* __restrict__ in, bf16_t* __restrict__ out, int n) {
  int i = blockIdx.x * blockDim.x + threadIdx.x;
  if (i < n) out[i] = (bf16_t)in[i];
}

// src == nullptr -> zeros (layer-2 h0); else copy f32 + bf16 shadow.
__global__ void __launch_bounds__(256)
init_h(const float* __restrict__ src, float* __restrict__ hf,
       bf16_t* __restrict__ hbf, int n) {
  int i = blockIdx.x * blockDim.x + threadIdx.x;
  if (i < n) {
    float v = src ? src[i] : 0.0f;
    hf[i] = v;
    hbf[i] = (bf16_t)v;
  }
}

__global__ void zero_u32x2(unsigned* p) {
  if (threadIdx.x < 2) p[threadIdx.x] = 0u;
}

// ---------------- input-projection GEMM ----------------------------------

struct GemmFrags {
  bf16x16 a0, a1, b[4];
};

__device__ __forceinline__ void
gemm_load(GemmFrags& f, const bf16_t* ap0, const bf16_t* ap1,
          const bf16_t* bp, int k0) {
  f.a0 = load_a_frag(ap0 + k0);
  f.a1 = load_a_frag(ap1 + k0);
#pragma unroll
  for (int t = 0; t < 4; ++t)
    f.b[t] = *(const bf16x16*)(bp + (size_t)t * 16 * F_N + k0);
}

__device__ __forceinline__ void
gemm_compute(f32x8 (&acc)[2][4], const GemmFrags& f) {
#pragma unroll
  for (int t = 0; t < 4; ++t) {
    acc[0][t] = wmma_bf16(f.a0, f.b[t], acc[0][t]);
    acc[1][t] = wmma_bf16(f.a1, f.b[t], acc[1][t]);
  }
}

// XP[M,3072] = A[M,1024](bf16) @ W + bias. Block 256 thr = 8 waves in 4x2;
// block tile 128x128, wave tile 32x64 (8 accumulators). Even/odd register-
// set software pipeline: no rotate-copies, one load-group of prefetch.
__global__ void __launch_bounds__(256)
gemm_xp(const bf16_t* __restrict__ A, const bf16_t* __restrict__ Bt,
        const float* __restrict__ bias, float* __restrict__ XP) {
  const int K = F_N, N = G_N;
  int wave = threadIdx.x >> 5;
  int lane = threadIdx.x & 31;
  int half = lane >> 4, l = lane & 15;
  int wr = wave & 3, wc = wave >> 2;            // 4 (M) x 2 (N) waves
  int m0 = blockIdx.y * 128 + wr * 32;
  int n0 = blockIdx.x * 128 + wc * 64;

  f32x8 acc[2][4] = {};
  const bf16_t* ap0 = A + (size_t)(m0 + l) * K + half * 16;
  const bf16_t* ap1 = ap0 + (size_t)16 * K;
  const bf16_t* bp  = Bt + (size_t)(n0 + l) * K + half * 16;

  GemmFrags C, Nx;
  gemm_load(C, ap0, ap1, bp, 0);
#pragma unroll 1
  for (int k0 = 0; k0 < K - 64; k0 += 64) {
    gemm_load(Nx, ap0, ap1, bp, k0 + 32);
    gemm_compute(acc, C);
    gemm_load(C, ap0, ap1, bp, k0 + 64);
    gemm_compute(acc, Nx);
  }
  gemm_load(Nx, ap0, ap1, bp, K - 32);
  gemm_compute(acc, C);     // step K-64
  gemm_compute(acc, Nx);    // step K-32

  // C layout: VGPR r -> M = r (lanes 0-15) / r+8 (lanes 16-31); N = lane&15.
#pragma unroll
  for (int s = 0; s < 2; ++s) {
#pragma unroll
    for (int t = 0; t < 4; ++t) {
      int col = n0 + t * 16 + l;
      float bv = bias[col];
#pragma unroll
      for (int r = 0; r < 8; ++r) {
        int row = m0 + s * 16 + r + half * 8;
        XP[(size_t)row * N + col] = acc[s][t][r] + bv;
      }
    }
  }
}

// ---------------- persistent recurrent scan -------------------------------

struct ScanFrags {
  bf16x16 a, bz, br, bh;
};

__device__ __forceinline__ void
scan_load(ScanFrags& f, const bf16_t* ap, const bf16_t* lz,
          const bf16_t* lr, const bf16_t* lh, int k0) {
  f.a  = load_a_frag(ap + k0);
  f.bz = *(const bf16x16*)(lz + k0);
  f.br = *(const bf16x16*)(lr + k0);
  f.bh = *(const bf16x16*)(lh + k0);
}

__device__ __forceinline__ void
scan_compute(f32x8& az, f32x8& ar, f32x8& ah, const ScanFrags& f) {
  az = wmma_bf16(f.a, f.bz, az);
  ar = wmma_bf16(f.a, f.br, ar);
  ah = wmma_bf16(f.a, f.bh, ah);
}

// One launch per layer. Block j owns units [16j,16j+16): stages its 48x1024
// bf16 U-slice (z/r/h gate columns) into LDS once, then loops t=0..255:
// 3-gate WMMA GEMM (A from global h ping-pong, B from LDS), gate math in
// registers, h/y stores, then a device-scope atomic grid barrier.
__global__ void __launch_bounds__(128)
gru_scan(const bf16_t* __restrict__ Ut, const float* __restrict__ rbias,
         const float* __restrict__ XP,
         bf16_t* hbf0, bf16_t* hbf1, float* hf0, float* hf1,
         bf16_t* ybf,              // layer1: bf16 y for next GEMM (or null)
         float* yf,                // layer2: f32 output (or null)
         float* state,             // layer2 final state @ t==T-1 (or null)
         unsigned* bar) {
  extern __shared__ bf16_t Us[];   // 48 rows x 1024 = 96KB
  const int K = U_N;
  const int tid = threadIdx.x;
  int j16 = blockIdx.x * 16;

  // cooperative stage of U-slice: rows 0-15 = z cols, 16-31 = r, 32-47 = h
  for (int i = tid; i < 48 * 128; i += 128) {      // 16B chunks
    int row = i >> 7, c = i & 127;
    int g = (row >> 4) * 1024 + j16 + (row & 15);  // global Ut row
    *(uint4*)(Us + (size_t)row * K + c * 8) =
        *(const uint4*)(Ut + (size_t)g * K + c * 8);
  }
  __syncthreads();

  int wave = tid >> 5, lane = tid & 31;
  int half = lane >> 4, l = lane & 15;
  const bf16_t* lz = Us + (size_t)(l) * K      + half * 16;
  const bf16_t* lr = Us + (size_t)(16 + l) * K + half * 16;
  const bf16_t* lh = Us + (size_t)(32 + l) * K + half * 16;
  int u = j16 + l;
  float rbz = rbias[u], rbr = rbias[1024 + u], rbh = rbias[2048 + u];

  for (int t = 0; t < T_N; ++t) {
    const bf16_t* hin_bf = (t & 1) ? hbf1 : hbf0;
    const float*  hin_f  = (t & 1) ? hf1  : hf0;
    bf16_t*       hob    = (t & 1) ? hbf0 : hbf1;
    float*        hof    = (t & 1) ? hf0  : hf1;

    f32x8 az = {}, ar = {}, ah = {};
    const bf16_t* ap = hin_bf + (size_t)(wave * 16 + l) * K + half * 16;

    ScanFrags C, Nx;
    scan_load(C, ap, lz, lr, lh, 0);
#pragma unroll 1
    for (int k0 = 0; k0 < K - 64; k0 += 64) {
      scan_load(Nx, ap, lz, lr, lh, k0 + 32);
      scan_compute(az, ar, ah, C);
      scan_load(C, ap, lz, lr, lh, k0 + 64);
      scan_compute(az, ar, ah, Nx);
    }
    scan_load(Nx, ap, lz, lr, lh, K - 32);
    scan_compute(az, ar, ah, C);
    scan_compute(az, ar, ah, Nx);

#pragma unroll
    for (int r = 0; r < 8; ++r) {
      int b = wave * 16 + r + half * 8;
      size_t xrow = ((size_t)b * T_N + t) * (size_t)G_N;
      float xz = XP[xrow + u];
      float xr = XP[xrow + 1024 + u];
      float xh = XP[xrow + 2048 + u];
      float z  = sigmoidf(xz + az[r] + rbz);
      float rg = sigmoidf(xr + ar[r] + rbr);
      float hh = tanhf(xh + rg * (ah[r] + rbh));
      float hold = hin_f[(size_t)b * U_N + u];
      float hn = z * hold + (1.0f - z) * hh;
      hof[(size_t)b * U_N + u] = hn;
      hob[(size_t)b * U_N + u] = (bf16_t)hn;
      if (ybf) ybf[((size_t)b * T_N + t) * U_N + u] = (bf16_t)hn;
      if (yf)  yf[((size_t)b * T_N + t) * U_N + u] = hn;
      if (state && t == T_N - 1) state[(size_t)b * U_N + u] = hn;
    }

    // device-wide step barrier (release arrive / acquire spin)
    __threadfence();
    __syncthreads();
    if (tid == 0) {
      __hip_atomic_fetch_add(bar, 1u, __ATOMIC_RELEASE,
                             __HIP_MEMORY_SCOPE_AGENT);
      unsigned target = (unsigned)(t + 1) * gridDim.x;
      while (__hip_atomic_load(bar, __ATOMIC_ACQUIRE,
                               __HIP_MEMORY_SCOPE_AGENT) < target)
        __builtin_amdgcn_s_sleep(1);
    }
    __syncthreads();
  }
}

// ---------------- host orchestration --------------------------------------

extern "C" void kernel_launch(void* const* d_in, const int* in_sizes, int n_in,
                              void* d_out, int out_size, void* d_ws,
                              size_t ws_size, hipStream_t stream) {
  const float* x      = (const float*)d_in[0];
  const float* hidden = (const float*)d_in[1];
  const float* W1     = (const float*)d_in[2];
  const float* U1     = (const float*)d_in[3];
  const float* b1     = (const float*)d_in[4];   // [2,3072]
  const float* W2     = (const float*)d_in[5];
  const float* U2     = (const float*)d_in[6];
  const float* b2     = (const float*)d_in[7];
  (void)in_sizes; (void)n_in; (void)out_size; (void)ws_size;

  float* out   = (float*)d_out;                        // [B,T,U]
  float* state = out + (size_t)B_N * T_N * U_N;        // [B,U]

  char* ws = (char*)d_ws;
  size_t off = 0;
  auto alloc = [&](size_t bytes) -> void* {
    void* p = ws + off;
    off = (off + bytes + 255) & ~(size_t)255;
    return p;
  };
  bf16_t*   xbf  = (bf16_t*)alloc((size_t)B_N * T_N * F_N * 2);
  bf16_t*   W1t  = (bf16_t*)alloc((size_t)G_N * F_N * 2);
  bf16_t*   U1t  = (bf16_t*)alloc((size_t)G_N * U_N * 2);
  bf16_t*   W2t  = (bf16_t*)alloc((size_t)G_N * U_N * 2);
  bf16_t*   U2t  = (bf16_t*)alloc((size_t)G_N * U_N * 2);
  float*    XP   = (float*) alloc((size_t)B_N * T_N * G_N * 4);  // reused
  bf16_t*   y1bf = (bf16_t*)alloc((size_t)B_N * T_N * U_N * 2);
  bf16_t*   hbf0 = (bf16_t*)alloc((size_t)B_N * U_N * 2);
  bf16_t*   hbf1 = (bf16_t*)alloc((size_t)B_N * U_N * 2);
  float*    hf0  = (float*) alloc((size_t)B_N * U_N * 4);
  float*    hf1  = (float*) alloc((size_t)B_N * U_N * 4);
  unsigned* bars = (unsigned*)alloc(256);              // [0]=layer1 [1]=layer2

  // 1) barrier counters must be zero every call (graph replays!)
  zero_u32x2<<<1, 64, 0, stream>>>(bars);

  // 2) weight transpose + bf16 convert
  dim3 tb(32, 8);
  dim3 tg(G_N / 32, F_N / 32);
  transpose_cvt<<<tg, tb, 0, stream>>>(W1, W1t, F_N, G_N);
  transpose_cvt<<<tg, tb, 0, stream>>>(U1, U1t, U_N, G_N);
  transpose_cvt<<<tg, tb, 0, stream>>>(W2, W2t, U_N, G_N);
  transpose_cvt<<<tg, tb, 0, stream>>>(U2, U2t, U_N, G_N);

  // 3) x -> bf16
  int nX = B_N * T_N * F_N;
  cvt_bf16<<<(nX + 255) / 256, 256, 0, stream>>>(x, xbf, nX);

  // 4) layer-1 h0 from provided `hidden`
  int nH = B_N * U_N;
  init_h<<<(nH + 255) / 256, 256, 0, stream>>>(hidden, hf0, hbf0, nH);

  // 5) XP1 = x @ W1 + b1[0]
  dim3 gg(G_N / 128, (B_N * T_N) / 128);
  gemm_xp<<<gg, 256, 0, stream>>>(xbf, W1t, b1, XP);

  // 6) layer-1 persistent scan (one launch, 256 internal steps)
  gru_scan<<<64, 128, 48 * 1024 * sizeof(bf16_t), stream>>>(
      U1t, b1 + G_N, XP, hbf0, hbf1, hf0, hf1,
      y1bf, nullptr, nullptr, bars + 0);

  // 7) XP2 = y1 @ W2 + b2[0]  (XP reused)
  gemm_xp<<<gg, 256, 0, stream>>>(y1bf, W2t, b2, XP);

  // 8) layer-2 h0 = zeros
  init_h<<<(nH + 255) / 256, 256, 0, stream>>>(nullptr, hf0, hbf0, nH);

  // 9) layer-2 persistent scan -> d_out (+ final state)
  gru_scan<<<64, 128, 48 * 1024 * sizeof(bf16_t), stream>>>(
      U2t, b2 + G_N, XP, hbf0, hbf1, hf0, hf1,
      nullptr, out, state, bars + 1);
}
